// Attention_34342558499101
// MI455X (gfx1250) — matrix operand order repaired and checked by reading, not compiled, not run on hardware
//
#include <hip/hip_runtime.h>
#include <math.h>

// Problem constants (from reference): B=32, T=2048, H=512, D=2H=1024
#define BB   32
#define TT   2048
#define HH   512
#define DD   1024

typedef __attribute__((ext_vector_type(16))) __bf16 v16bf;
typedef __attribute__((ext_vector_type(8)))  __bf16 v8bf;
typedef __attribute__((ext_vector_type(8)))  float  v8f;

union AFrag { v16bf v; v8bf h[2]; };

__device__ __forceinline__ unsigned short f2bf(float f) {
    unsigned u = __float_as_uint(f);
    u += 0x7FFFu + ((u >> 16) & 1u);           // RNE (prep kernels, cheap path not needed)
    return (unsigned short)(u >> 16);
}

// pack two f32 -> two bf16 in one u32 with one v_perm_b32
// (round-to-nearest via +0x8000, then take high halves)
__device__ __forceinline__ unsigned pack2bf(float lo, float hi) {
    unsigned a = __float_as_uint(lo) + 0x8000u;
    unsigned b = __float_as_uint(hi) + 0x8000u;
    // v_perm_b32: byte sel 0-3 from 2nd operand(low), 4-7 from 1st (high)
    // out = { hi16(lo), hi16(hi) }  -> sel bytes {2,3,6,7}
    return __builtin_amdgcn_perm(b, a, 0x07060302u);
}

// ---------------------------------------------------------------------------
// Phase 0a: hid_proj[b,h] = hidden[b,:] . Wh[h,:] + b_attn[h]   (tiny GEMV)
// ---------------------------------------------------------------------------
__global__ void prep_hid_kernel(const float* __restrict__ hidden,
                                const float* __restrict__ W,
                                const float* __restrict__ battn,
                                float* __restrict__ hp) {
    const int h = blockIdx.x * 64 + threadIdx.x;
    const int b = blockIdx.y;
    const float* hrow = hidden + (size_t)b * DD;
    const float* wrow = W + (size_t)h * (2 * DD);      // Wh = W[:, :D]
    float s = 0.f;
    #pragma unroll 4
    for (int d = 0; d < DD; d += 4) {
        float4 a = *reinterpret_cast<const float4*>(hrow + d);
        float4 w = *reinterpret_cast<const float4*>(wrow + d);
        s += a.x * w.x + a.y * w.y + a.z * w.z + a.w * w.w;
    }
    hp[b * HH + h] = s + battn[h];
}

// ---------------------------------------------------------------------------
// Phase 0b: We (H x D, = W[:, D:]) -> bf16 row-major in workspace
// ---------------------------------------------------------------------------
__global__ void prep_we_kernel(const float* __restrict__ W,
                               unsigned short* __restrict__ webf) {
    const int i = blockIdx.x * 256 + threadIdx.x;   // i < H*D
    const int h = i >> 10;
    const int d = i & (DD - 1);
    webf[i] = f2bf(W[(size_t)h * (2 * DD) + DD + d]);
}

// ---------------------------------------------------------------------------
// Phase 1: scores[b,t] = v . tanh( enc[b,t,:] @ We^T + hid_proj[b,:] )
// WG = 64 T-rows x 512 H-cols of one batch; K chunked by 128 through LDS.
// Each wave: 4 row-strips x 4 col-tiles (64 cols), 4 A-frags x 4 B-frags
// -> 16 WMMAs per k-step with only 8 fragment loads (4x less B traffic).
// ---------------------------------------------------------------------------
__global__ void
__launch_bounds__(256)
attn_main_kernel(const float* __restrict__ enc,
                 const unsigned short* __restrict__ webf,
                 const float* __restrict__ hp,
                 const float* __restrict__ vvec,
                 float* __restrict__ sc) {
    __shared__ unsigned short lA[64 * 136];  // 64 rows x 128 K bf16, +8 pad
    __shared__ float lS[64];

    const int tid  = threadIdx.x;
    const int wave = tid >> 5;
    const int lane = tid & 31;
    const int ln   = lane & 15;
    const int hi   = lane >> 4;
    const int h0   = wave * 64;         // this wave's 64-column H slice

    const int b  = blockIdx.y;
    const int t0 = blockIdx.x * 64;

    if (tid < 64) lS[tid] = 0.f;

    v8f acc[4][4];                      // [strip][coltile]
    #pragma unroll
    for (int s = 0; s < 4; ++s)
        #pragma unroll
        for (int t = 0; t < 4; ++t)
            #pragma unroll
            for (int r = 0; r < 8; ++r) acc[s][t][r] = 0.f;

    const int sr = tid >> 2;            // staging: row 0..63
    const int sq = tid & 3;             // 32-col quad

    for (int kc = 0; kc < DD / 128; ++kc) {
        __syncthreads();
        {
            const float* src = enc + ((size_t)b * TT + (size_t)(t0 + sr)) * DD
                                   + (size_t)(kc * 128 + sq * 32);
            unsigned short* dst = lA + sr * 136 + sq * 32;
            #pragma unroll
            for (int i = 0; i < 32; i += 8) {
                float4 x = *reinterpret_cast<const float4*>(src + i);
                float4 y = *reinterpret_cast<const float4*>(src + i + 4);
                uint4 o;
                o.x = pack2bf(x.x, x.y);
                o.y = pack2bf(x.z, x.w);
                o.z = pack2bf(y.x, y.y);
                o.w = pack2bf(y.z, y.w);
                *reinterpret_cast<uint4*>(dst + i) = o;
            }
        }
        __syncthreads();

        #pragma unroll
        for (int ks = 0; ks < 4; ++ks) {
            const int kL = ks * 32;
            // 4 A fragments, one per 16-row strip
            AFrag a[4];
            #pragma unroll
            for (int s = 0; s < 4; ++s) {
                const unsigned short* ap =
                    lA + (s * 16 + ln) * 136 + kL + hi * 16;
                a[s].h[0] = *reinterpret_cast<const v8bf*>(ap);
                a[s].h[1] = *reinterpret_cast<const v8bf*>(ap + 8);
            }
            // 4 B fragments, one per 16-col tile of this wave's slice
            const int kg = kc * 128 + kL + hi * 16;
            v16bf bf[4];
            #pragma unroll
            for (int t = 0; t < 4; ++t) {
                const int h = h0 + t * 16 + ln;
                bf[t] = *reinterpret_cast<const v16bf*>(webf + (size_t)h * DD + kg);
            }
            #pragma unroll
            for (int s = 0; s < 4; ++s)
                #pragma unroll
                for (int t = 0; t < 4; ++t)
                    acc[s][t] = __builtin_amdgcn_wmma_f32_16x16x32_bf16(
                        false, a[s].v, false, bf[t], (short)0, acc[s][t],
                        false, false);
        }
    }

    // Epilogue: P[s][r] = sum over this wave's 64 cols of v[h]*tanh(acc+hp)
    // C layout: VGPR r -> row r + hi*8 of strip s, col = ln of tile t.
    float P[4][8];
    #pragma unroll
    for (int s = 0; s < 4; ++s)
        #pragma unroll
        for (int r = 0; r < 8; ++r) P[s][r] = 0.f;

    #pragma unroll
    for (int t = 0; t < 4; ++t) {
        const int h = h0 + t * 16 + ln;
        const float hb = hp[b * HH + h];
        const float vv = vvec[h];
        #pragma unroll
        for (int s = 0; s < 4; ++s)
            #pragma unroll
            for (int r = 0; r < 8; ++r)
                P[s][r] += vv * tanhf(acc[s][t][r] + hb);
    }
    #pragma unroll
    for (int s = 0; s < 4; ++s)
        #pragma unroll
        for (int r = 0; r < 8; ++r) {
            P[s][r] += __shfl_xor(P[s][r], 8);
            P[s][r] += __shfl_xor(P[s][r], 4);
            P[s][r] += __shfl_xor(P[s][r], 2);
            P[s][r] += __shfl_xor(P[s][r], 1);
        }
    if (ln == 0) {
        #pragma unroll
        for (int s = 0; s < 4; ++s)
            #pragma unroll
            for (int r = 0; r < 8; ++r)
                atomicAdd(&lS[s * 16 + hi * 8 + r], P[s][r]);
    }
    __syncthreads();
    if (tid < 64) sc[(size_t)b * TT + t0 + tid] = lS[tid];
}

// ---------------------------------------------------------------------------
// Phase 2: softmax over T per batch. grid B, block 256, 8 elems/thread.
// ---------------------------------------------------------------------------
__global__ void softmax_kernel(const float* __restrict__ sc,
                               float* __restrict__ out) {
    __shared__ float red[256];
    const int b = blockIdx.x, tid = threadIdx.x;
    float x[8];
    float m = -1e30f;
    #pragma unroll
    for (int i = 0; i < 8; ++i) {
        x[i] = sc[(size_t)b * TT + tid + i * 256];
        m = fmaxf(m, x[i]);
    }
    red[tid] = m; __syncthreads();
    for (int s = 128; s > 0; s >>= 1) {
        if (tid < s) red[tid] = fmaxf(red[tid], red[tid + s]);
        __syncthreads();
    }
    m = red[0]; __syncthreads();
    float s = 0.f;
    #pragma unroll
    for (int i = 0; i < 8; ++i) { x[i] = __expf(x[i] - m); s += x[i]; }
    red[tid] = s; __syncthreads();
    for (int s2 = 128; s2 > 0; s2 >>= 1) {
        if (tid < s2) red[tid] += red[tid + s2];
        __syncthreads();
    }
    const float inv = 1.f / red[0];
    #pragma unroll
    for (int i = 0; i < 8; ++i)
        out[(size_t)b * TT + tid + i * 256] = x[i] * inv;
}

// ---------------------------------------------------------------------------
extern "C" void kernel_launch(void* const* d_in, const int* in_sizes, int n_in,
                              void* d_out, int out_size, void* d_ws, size_t ws_size,
                              hipStream_t stream) {
    const float* hidden = (const float*)d_in[0];   // (B, D)
    const float* enc    = (const float*)d_in[1];   // (B, T, D)
    const float* W      = (const float*)d_in[2];   // (H, 2D)
    const float* battn  = (const float*)d_in[3];   // (H,)
    const float* v      = (const float*)d_in[4];   // (H,)
    float* out = (float*)d_out;                    // (B, 1, T)

    char* ws = (char*)d_ws;
    float*          hp   = (float*)ws;                              // B*H f32   = 64 KB
    unsigned short* webf = (unsigned short*)(ws + 65536);           // H*D bf16  = 1 MB
    float*          sc   = (float*)(ws + 65536 + 1048576);          // B*T f32   = 256 KB

    prep_hid_kernel<<<dim3(HH / 64, BB), 64, 0, stream>>>(hidden, W, battn, hp);
    prep_we_kernel<<<(HH * DD) / 256, 256, 0, stream>>>(W, webf);
    attn_main_kernel<<<dim3(TT / 64, BB), 256, 0, stream>>>(enc, webf, hp, v, sc);
    softmax_kernel<<<BB, 256, 0, stream>>>(sc, out);
}
